// AdaArcLoss_86105504350826
// MI455X (gfx1250) — matrix operand order repaired and checked by reading, not compiled, not run on hardware
//
#include <hip/hip_runtime.h>
#include <hip/hip_bf16.h>
#include <math.h>

typedef __bf16 bf16;
typedef __attribute__((ext_vector_type(16))) __bf16 v16bf;
typedef __attribute__((ext_vector_type(4)))  __bf16 bf16x4;
typedef __attribute__((ext_vector_type(8)))  float v8f;
typedef __attribute__((ext_vector_type(4)))  float f32x4;
typedef __attribute__((ext_vector_type(4)))  unsigned int u32x4;

#define EMB      512
#define BATCH    256
#define NCLS     100000
#define SCALE_   32.0f
#define SHIFT_   48.0f
#define SINM_    0.4794255386042030f   // sin(0.5)
#define NT       32                    // classes per tile (100000 % 32 == 0)
#define NTILES   (NCLS / NT)           // 3125
#define LSTRIDE  520                   // padded row stride in bf16 (1040 B, 16B aligned)
#define A_BYTES  (BATCH * LSTRIDE * 2) // 266240
#define B_BYTES  (NT * LSTRIDE * 2)    // 33280
#define SMEM_BYTES (A_BYTES + B_BYTES + NT * 4)
#define K4_GRID  256

// ---------- helpers ----------
__device__ __forceinline__ float redsum8(float v) {
  v += __shfl_xor(v, 1, 32); v += __shfl_xor(v, 2, 32); v += __shfl_xor(v, 4, 32);
  return v;
}
__device__ __forceinline__ float redsum16(float v) {
  v += __shfl_xor(v, 1, 32); v += __shfl_xor(v, 2, 32);
  v += __shfl_xor(v, 4, 32); v += __shfl_xor(v, 8, 32);
  return v;
}
__device__ __forceinline__ float redsum32(float v) {
  v = redsum16(v); v += __shfl_xor(v, 16, 32); return v;
}
__device__ __forceinline__ v16bf ld_frag(const char* p0, const char* p1) {
  union { v16bf v; u32x4 q[2]; } u;
  u.q[0] = *(const u32x4*)p0;
  u.q[1] = *(const u32x4*)p1;
  return u.v;
}

// ---------- K1: normalize features, emit f32 fn + bf16 A (row major) ----------
__global__ void k1_norm_feat(const float* __restrict__ feat,
                             float* __restrict__ fn, bf16* __restrict__ Ag) {
  int row = blockIdx.x * 8 + (threadIdx.x >> 5);
  int l = threadIdx.x & 31;
  const float* src = feat + row * EMB + l * 16;
  f32x4 v[4];
  float ss = 0.f;
#pragma unroll
  for (int i = 0; i < 4; i++) {
    v[i] = *(const f32x4*)(src + 4 * i);
    ss += v[i].x * v[i].x + v[i].y * v[i].y + v[i].z * v[i].z + v[i].w * v[i].w;
  }
  ss = redsum32(ss);
  float sc = 1.0f / fmaxf(sqrtf(ss), 1e-12f);
  float* dn = fn + row * EMB + l * 16;
  bf16* db = Ag + row * EMB + l * 16;
#pragma unroll
  for (int i = 0; i < 4; i++) {
    f32x4 w; w.x = v[i].x * sc; w.y = v[i].y * sc; w.z = v[i].z * sc; w.w = v[i].w * sc;
    *(f32x4*)(dn + 4 * i) = w;
    bf16x4 b; b.x = (bf16)w.x; b.y = (bf16)w.y; b.z = (bf16)w.z; b.w = (bf16)w.w;
    *(bf16x4*)(db + 4 * i) = b;
  }
}

// ---------- K2: exact cos_y per row via gathered label row ----------
__global__ void k2_cosy(const float* __restrict__ fn, const float* __restrict__ W,
                        const int* __restrict__ labels,
                        float* __restrict__ cosy, float* __restrict__ diff) {
  int row = blockIdx.x * 8 + (threadIdx.x >> 5);
  int l = threadIdx.x & 31;
  long long lab = labels[row];
  const float* w = W + lab * (long long)EMB + l * 16;
  const float* f = fn + row * EMB + l * 16;
  float dot = 0.f, ss = 0.f;
#pragma unroll
  for (int i = 0; i < 16; i++) { float wv = w[i], fv = f[i]; dot += wv * fv; ss += wv * wv; }
  dot = redsum32(dot); ss = redsum32(ss);
  if (l == 0) {
    float c = dot / fmaxf(sqrtf(ss), 1e-12f);
    c = fminf(fmaxf(c, -1.f), 1.f);
    cosy[row] = c; diff[row] = 1.f - c;
  }
}

// ---------- K3: quantile threshold, margins, target logits, zero partials ----------
__global__ void k3_margin(const float* __restrict__ cosy, const float* __restrict__ diff,
                          float* __restrict__ tlog, float* __restrict__ delta,
                          float* __restrict__ partial) {
  __shared__ float sd[BATCH];
  __shared__ float sthr;
  int i = threadIdx.x;
  float d = diff[i];
  sd[i] = d;
  __syncthreads();
  int rank = 0;
  for (int j = 0; j < BATCH; j++) {
    float dj = sd[j];
    rank += (dj < d) || (dj == d && j < i);
  }
  if (rank == 51) sthr = d;  // quantile(0.2) index = 0.2*255 = 51 exactly
  __syncthreads();
  float thr = sthr;
  float m = 0.5f + 0.1f * d + ((d >= thr) ? 0.15f : 0.f);
  float cy = cosy[i];
  float tl = SCALE_ * (cy * cosf(m * (float)M_PI) - m * SINM_);
  tlog[i] = tl;
  delta[i] = __expf(tl - SHIFT_) - __expf(SCALE_ * cy - SHIFT_);
  partial[i] = 0.f;
}

// ---------- K4: fused bf16-WMMA GEMM + on-the-fly W-normalization + sum-exp ----------
__global__ void __launch_bounds__(256)
k4_main(const float* __restrict__ W, const bf16* __restrict__ Ag,
        float* __restrict__ partial) {
  extern __shared__ __align__(16) char smem[];
  bf16* Bld = (bf16*)(smem + A_BYTES);
  float* rcpn = (float*)(smem + A_BYTES + B_BYTES);

  const int tid = threadIdx.x;
  const int l = tid & 31;
  const int w = tid >> 5;

  // Stage A (256x512 bf16) into LDS with padded stride: one row per thread.
  {
    const u32x4* src = (const u32x4*)(Ag + tid * EMB);
    char* dst = smem + (size_t)tid * (LSTRIDE * 2);
#pragma unroll 8
    for (int j = 0; j < 64; j++) *(u32x4*)(dst + j * 16) = src[j];
  }

  v8f acc0 = {}, acc1 = {};     // running exp-sums: rows 32w..32w+15 / +16..+31
  f32x4 rg[16];                 // register staging (64 f32/thread)

  auto load_regs = [&](int tile) {
    long long cls = (long long)tile * NT + (tid >> 3);
    const f32x4* src = (const f32x4*)(W + cls * EMB + (tid & 7) * 64);
#pragma unroll
    for (int k = 0; k < 16; k++) rg[k] = src[k];
  };
  auto store_tile = [&]() {
    float ss = 0.f;
    int crow = tid >> 3, seg = tid & 7;
    char* dst = (char*)Bld + crow * (LSTRIDE * 2) + seg * 128;
#pragma unroll
    for (int k = 0; k < 16; k++) {
      f32x4 v = rg[k];
      ss += v.x * v.x + v.y * v.y + v.z * v.z + v.w * v.w;
      bf16x4 b; b.x = (bf16)v.x; b.y = (bf16)v.y; b.z = (bf16)v.z; b.w = (bf16)v.w;
      *(bf16x4*)(dst + k * 8) = b;
    }
    ss = redsum8(ss);  // 8 threads per class
    if ((tid & 7) == 0) rcpn[crow] = 1.0f / fmaxf(sqrtf(ss), 1e-12f);
  };

  const int first = blockIdx.x;
  if (first < NTILES) { load_regs(first); store_tile(); }

  // Fragment base addresses (contiguous k order, same for A and B):
  //  A (16x32 bf16): lane<16 -> K{0..15} (32B) ; lane>=16 -> K{16..31}
  //  B (32x16 bf16): lane<16 -> K{0..15} (32B) ; lane>=16 -> K{16..31}
  const char* Ab0 = smem + (size_t)(w * 32 + (l & 15)) * (LSTRIDE * 2) + ((l >> 4) * 32);
  const char* Ab1 = Ab0 + 16 * (LSTRIDE * 2);
  const char* Bb0 = (const char*)Bld + (size_t)(l & 15) * (LSTRIDE * 2) + ((l >> 4) * 32);
  const char* Bb1 = Bb0 + 16 * (LSTRIDE * 2);

  for (int tile = first; tile < NTILES; tile += gridDim.x) {
    __syncthreads();                       // B tile + rcpn + (first iter) A ready
    int nxt = tile + (int)gridDim.x;
    bool hasNext = nxt < NTILES;
    if (hasNext) load_regs(nxt);           // overlap global loads with WMMA below

    v8f c00 = {}, c01 = {}, c10 = {}, c11 = {};
#pragma unroll
    for (int ks = 0; ks < 16; ks++) {
      int o = ks * 64;
      v16bf a0 = ld_frag(Ab0 + o, Ab0 + o + 16);
      v16bf a1 = ld_frag(Ab1 + o, Ab1 + o + 16);
      v16bf b0 = ld_frag(Bb0 + o, Bb0 + o + 16);
      v16bf b1 = ld_frag(Bb1 + o, Bb1 + o + 16);
      c00 = __builtin_amdgcn_wmma_f32_16x16x32_bf16(false, a0, false, b0, (short)0, c00, false, false);
      c01 = __builtin_amdgcn_wmma_f32_16x16x32_bf16(false, a0, false, b1, (short)0, c01, false, false);
      c10 = __builtin_amdgcn_wmma_f32_16x16x32_bf16(false, a1, false, b0, (short)0, c10, false, false);
      c11 = __builtin_amdgcn_wmma_f32_16x16x32_bf16(false, a1, false, b1, (short)0, c11, false, false);
    }

    float r0 = rcpn[l & 15];        // N-subtile 0 class norms
    float r1 = rcpn[16 + (l & 15)]; // N-subtile 1
#pragma unroll
    for (int r = 0; r < 8; r++) {
      float x;
      x = fminf(fmaxf(c00[r] * r0, -1.f), 1.f); acc0[r] += __expf(fmaf(SCALE_, x, -SHIFT_));
      x = fminf(fmaxf(c01[r] * r1, -1.f), 1.f); acc0[r] += __expf(fmaf(SCALE_, x, -SHIFT_));
      x = fminf(fmaxf(c10[r] * r0, -1.f), 1.f); acc1[r] += __expf(fmaf(SCALE_, x, -SHIFT_));
      x = fminf(fmaxf(c11[r] * r1, -1.f), 1.f); acc1[r] += __expf(fmaf(SCALE_, x, -SHIFT_));
    }

    if (hasNext) { __syncthreads(); store_tile(); }
  }

  // Reduce exp-sums over the 16 classes held per 16-lane half, then accumulate.
#pragma unroll
  for (int r = 0; r < 8; r++) {
    float v0 = redsum16(acc0[r]);
    float v1 = redsum16(acc1[r]);
    if ((l & 15) == 0) {
      int half = (l >> 4) * 8;  // lane0 -> rows r, lane16 -> rows r+8
      atomicAdd(&partial[w * 32 + r + half], v0);
      atomicAdd(&partial[w * 32 + 16 + r + half], v1);
    }
  }
}

// ---------- K5: per-row LSE, mean CE loss ----------
__global__ void k5_final(const float* __restrict__ partial, const float* __restrict__ delta,
                         const float* __restrict__ tlog, float* __restrict__ out) {
  __shared__ float red[8];
  int i = threadIdx.x;
  float s = partial[i] + delta[i];   // swap raw label term for margin term
  float li = (SHIFT_ + __logf(s)) - tlog[i];
  float v = redsum32(li);
  if ((i & 31) == 0) red[i >> 5] = v;
  __syncthreads();
  if (i == 0) {
    float t = 0.f;
    for (int k = 0; k < 8; k++) t += red[k];
    out[0] = t * (1.0f / BATCH);
  }
}

// ---------- launch ----------
extern "C" void kernel_launch(void* const* d_in, const int* in_sizes, int n_in,
                              void* d_out, int out_size, void* d_ws, size_t ws_size,
                              hipStream_t stream) {
  (void)in_sizes; (void)n_in; (void)out_size; (void)ws_size;
  const float* feat = (const float*)d_in[0];
  const int* labels = (const int*)d_in[1];
  const float* W = (const float*)d_in[2];
  float* out = (float*)d_out;

  char* ws = (char*)d_ws;
  float* fn      = (float*)ws;                                   // 256*512 f32
  bf16* Ag       = (bf16*)(ws + BATCH * EMB * 4);                // 256*512 bf16
  float* cosy    = (float*)(ws + BATCH * EMB * 4 + BATCH * EMB * 2);
  float* diff    = cosy + BATCH;
  float* tlog    = diff + BATCH;
  float* delta   = tlog + BATCH;
  float* partial = delta + BATCH;

  hipFuncSetAttribute(reinterpret_cast<const void*>(k4_main),
                      hipFuncAttributeMaxDynamicSharedMemorySize, SMEM_BYTES);

  k1_norm_feat<<<BATCH / 8, 256, 0, stream>>>(feat, fn, Ag);
  k2_cosy<<<BATCH / 8, 256, 0, stream>>>(fn, W, labels, cosy, diff);
  k3_margin<<<1, 256, 0, stream>>>(cosy, diff, tlog, delta, partial);
  k4_main<<<K4_GRID, 256, SMEM_BYTES, stream>>>(W, Ag, partial);
  k5_final<<<1, 256, 0, stream>>>(partial, delta, tlog, out);
}
